// SmallGCNEncoder_66133906423867
// MI455X (gfx1250) — matrix operand order repaired and checked by reading, not compiled, not run on hardware
//
#include <hip/hip_runtime.h>
#include <hip/hip_bf16.h>

// ---------------------------------------------------------------------------
// 2-layer GCN for MI455X (gfx1250, wave32).
//
// Roofline: GEMM1 streams x (102 MB) once -> ~4.4 us floor at 23.3 TB/s;
// 3.3 GFLOP is trivial next to that, so the GEMM is bandwidth-bound and we
// use bf16 WMMA (f32 accumulate) purely as the cheapest MAC path.
// Weights are pre-converted to bf16 and pre-swizzled into the WMMA
// B-fragment layout (one-time ~160 KB), so the GEMM inner loop is all
// b128 loads + v_wmma. The edge aggregation (2 x 800k edges x 256 B gather
// + 256 B of f32 atomics) dominates total runtime and is issued as
// 64 lanes/edge coalesced global_atomic_add_f32.
// ---------------------------------------------------------------------------

typedef __attribute__((ext_vector_type(16))) __bf16 v16bf;
typedef __attribute__((ext_vector_type(8)))  float  v8f;

#define WAVE 32

// ---------------- degree / norm kernels ----------------

__global__ void k_deg_init(float* __restrict__ deg, int n) {
    int i = blockIdx.x * blockDim.x + threadIdx.x;
    if (i < n) deg[i] = 1.0f;                      // self-loop contribution
}

__global__ void k_deg_accum(const int* __restrict__ dst, float* __restrict__ deg, int e) {
    int i = blockIdx.x * blockDim.x + threadIdx.x;
    if (i < e) atomicAdd(&deg[dst[i]], 1.0f);
}

__global__ void k_dinv(float* __restrict__ deg, int n) {
    int i = blockIdx.x * blockDim.x + threadIdx.x;
    if (i < n) deg[i] = rsqrtf(deg[i]);            // deg >= 1 always
}

__global__ void k_norm(const int* __restrict__ src, const int* __restrict__ dst,
                       const float* __restrict__ dinv, float* __restrict__ norm, int e) {
    int i = blockIdx.x * blockDim.x + threadIdx.x;
    if (i < e) norm[i] = dinv[src[i]] * dinv[dst[i]];
}

// ---------------- weight pre-swizzle into WMMA B-fragment layout ----------
//
// B (32x16 bf16) fragment layout (ISA 7.12.2): lane = (khi<<4)|(n&15) holds
// K = k0 + khi*8 + i (i=0..7), K = k0 + 16 + khi*8 + (i-8) (i=8..15), col n. We store fragment (kt, nt) for
// lane l as 16 contiguous bf16 at element offset ((kt*4+nt)*32 + l)*16,
// so the GEMM loads each lane's fragment with contiguous b128s.
__global__ void k_pack_w(const float* __restrict__ W, __bf16* __restrict__ Wp, int K) {
    int i = blockIdx.x * blockDim.x + threadIdx.x;   // over K*64 elements
    if (i >= K * 64) return;
    int k = i >> 6, n = i & 63;
    int kt = k >> 5, kr = k & 31;
    int khi = (kr >> 3) & 1, ki = (kr & 7) | ((kr >> 4) << 3);
    int nt = n >> 4;
    int lane = (khi << 4) | (n & 15);
    Wp[(size_t)(((kt * 4 + nt) * 32 + lane)) * 16 + ki] = (__bf16)W[i];
}

// ---------------- WMMA GEMM: Hout[M x 64] = (relu?)A[M x K] @ W[K x 64] ----
//
// One wave owns a 16x64 tile: 4 x v_wmma_f32_16x16x32_bf16 per 32-wide
// K-step. A fragment (16x32 bf16): lane 0-15 row=lane, elems 0..7 ->
// K=k0..k0+7, elems 8..15 -> K=k0+16..k0+23; lanes 16-31 shift K by 8.
// D (16x16 f32): VGPR v -> row v (lanes 0-15) / v+8 (lanes 16-31).
template <bool RELU>
__device__ __forceinline__ void cvt4(v16bf& a, int base, float4 f) {
    if (RELU) {
        f.x = fmaxf(f.x, 0.0f); f.y = fmaxf(f.y, 0.0f);
        f.z = fmaxf(f.z, 0.0f); f.w = fmaxf(f.w, 0.0f);
    }
    a[base + 0] = (__bf16)f.x; a[base + 1] = (__bf16)f.y;
    a[base + 2] = (__bf16)f.z; a[base + 3] = (__bf16)f.w;
}

template <bool RELU>
__global__ __launch_bounds__(256)
void k_gemm_wmma(const float* __restrict__ A, const __bf16* __restrict__ Wp,
                 float* __restrict__ Hout, int M, int K) {
    const int lane = threadIdx.x & (WAVE - 1);
    const int wave = threadIdx.x / WAVE;
    const int rt   = blockIdx.x * 8 + wave;        // 16-row tile index
    if (rt * 16 >= M) return;                      // wave-uniform: EXEC stays all-1s

    const int l15   = lane & 15;
    const int rhalf = lane >> 4;                   // 0 or 1
    const int arow  = rt * 16 + l15;

    const v16bf* bfrag = (const v16bf*)Wp;         // fragment-granular view
    v8f acc[4] = {};

    for (int k0 = 0, kt = 0; k0 < K; k0 += 32, ++kt) {
        const float* arowp = A + (size_t)arow * K + k0 + rhalf * 8;
        // prefetch the next k-slab of this lane's row (speculative, OOB-safe)
        __builtin_prefetch(arowp + 32, 0, 0);

        // ---- A fragment: two 32B-aligned runs of 8 floats -> 4x b128
        const float4* ap = (const float4*)arowp;
        v16bf a;
        cvt4<RELU>(a, 0,  ap[0]);   // K = +0..3
        cvt4<RELU>(a, 4,  ap[1]);   // K = +4..7
        cvt4<RELU>(a, 8,  ap[4]);   // K = +16..19
        cvt4<RELU>(a, 12, ap[5]);   // K = +20..23

        // ---- 4 pre-swizzled B fragments: 32 contiguous bytes per lane
#pragma unroll
        for (int nt = 0; nt < 4; ++nt) {
            v16bf b = bfrag[(size_t)(kt * 4 + nt) * 32 + lane];
            acc[nt] = __builtin_amdgcn_wmma_f32_16x16x32_bf16(
                /*neg_a=*/false, a, /*neg_b=*/false, b,
                /*c_mod=*/(short)0, acc[nt], /*reuse_a=*/false, /*reuse_b=*/false);
        }
    }

#pragma unroll
    for (int nt = 0; nt < 4; ++nt)
#pragma unroll
        for (int v = 0; v < 8; ++v) {
            int r = rt * 16 + v + rhalf * 8;
            Hout[(size_t)r * 64 + nt * 16 + l15] = acc[nt][v];
        }
}

// ---------------- aggregation ----------------

// agg[i,f] = bias[f] + h[i,f] * dinv[i]^2   (bias + self-loop edge, norm=dinv^2)
__global__ void k_agg_init(const float* __restrict__ h, const float* __restrict__ dinv,
                           const float* __restrict__ bias, float* __restrict__ agg, int n) {
    int i = blockIdx.x * blockDim.x + threadIdx.x;
    if (i >= n * 64) return;
    int node = i >> 6, f = i & 63;
    float di = dinv[node];
    agg[i] = bias[f] + h[i] * di * di;
}

// 64 threads per edge -> coalesced gather of h[src,:] and coalesced f32 atomics
__global__ void k_edge_scatter(const float* __restrict__ h, const int* __restrict__ src,
                               const int* __restrict__ dst, const float* __restrict__ norm,
                               float* __restrict__ agg, int e) {
    int tid = blockIdx.x * blockDim.x + threadIdx.x;
    if (tid >= e * 64) return;
    int ed = tid >> 6, f = tid & 63;
    float v = h[(size_t)src[ed] * 64 + f] * norm[ed];
    atomicAdd(&agg[(size_t)dst[ed] * 64 + f], v);
}

// ---------------- launch ----------------

extern "C" void kernel_launch(void* const* d_in, const int* in_sizes, int n_in,
                              void* d_out, int out_size, void* d_ws, size_t ws_size,
                              hipStream_t stream) {
    const float* x  = (const float*)d_in[0];
    const int*   ei = (const int*)  d_in[1];
    const float* W1 = (const float*)d_in[2];
    const float* b1 = (const float*)d_in[3];
    const float* W2 = (const float*)d_in[4];
    const float* b2 = (const float*)d_in[5];
    float* out = (float*)d_out;

    const int HID = in_sizes[3];            // 64
    const int IN  = in_sizes[2] / HID;      // 512
    const int N   = in_sizes[0] / IN;       // 50000
    const int E   = in_sizes[1] / 2;        // 800000

    const int* src = ei;
    const int* dst = ei + E;

    // workspace layout (float-granular slabs, 256-elem aligned)
    float* ws   = (float*)d_ws;
    size_t off  = 0;
    float* dinv = ws + off; off += (size_t)((N + 255) & ~255);
    float* norm = ws + off; off += (size_t)((E + 255) & ~255);
    float* h1   = ws + off; off += (size_t)N * 64;
    float* agg1 = ws + off; off += (size_t)N * 64;
    float* h2   = ws + off; off += (size_t)N * 64;
    __bf16* Wp1 = (__bf16*)(ws + off); off += (size_t)(IN * HID + 255) / 2 & ~(size_t)127;
    __bf16* Wp2 = (__bf16*)(ws + off);

    const int B = 256;
    auto cdiv = [](int a, int b) { return (a + b - 1) / b; };

    // norm precompute
    k_deg_init <<<cdiv(N, B), B, 0, stream>>>(dinv, N);
    k_deg_accum<<<cdiv(E, B), B, 0, stream>>>(dst, dinv, E);
    k_dinv     <<<cdiv(N, B), B, 0, stream>>>(dinv, N);
    k_norm     <<<cdiv(E, B), B, 0, stream>>>(src, dst, dinv, norm, E);

    // pre-swizzle weights into WMMA B-fragment layout (bf16)
    k_pack_w<<<cdiv(IN * HID, B), B, 0, stream>>>(W1, Wp1, IN);
    k_pack_w<<<cdiv(HID * HID, B), B, 0, stream>>>(W2, Wp2, HID);

    const int rowTiles = cdiv(N, 16);
    const int gemmGrid = cdiv(rowTiles, 8);    // 8 waves / block

    // layer 1: h1 = x @ W1 ; agg1 = b1 + selfloop + scatter
    k_gemm_wmma<false><<<gemmGrid, B, 0, stream>>>(x, Wp1, h1, N, IN);
    k_agg_init <<<cdiv(N * 64, B), B, 0, stream>>>(h1, dinv, b1, agg1, N);
    k_edge_scatter<<<cdiv(E * 64, B), B, 0, stream>>>(h1, src, dst, norm, agg1, E);

    // layer 2: h2 = relu(agg1) @ W2 ; out = b2 + selfloop + scatter
    k_gemm_wmma<true><<<gemmGrid, B, 0, stream>>>(agg1, Wp2, h2, N, HID);
    k_agg_init <<<cdiv(N * 64, B), B, 0, stream>>>(h2, dinv, b2, out, N);
    k_edge_scatter<<<cdiv(E * 64, B), B, 0, stream>>>(h2, src, dst, norm, out, E);
}
